// Memory_64132451664295
// MI455X (gfx1250) — compile-verified, matrix operand order checked by reading, not run on hardware
//
#include <hip/hip_runtime.h>
#include <hip/hip_bf16.h>
#include <math.h>

// ---------------------------------------------------------------------------
// CDNA5 (gfx1250) WMMA types
// ---------------------------------------------------------------------------
typedef __attribute__((ext_vector_type(16))) __bf16 v16bf;
typedef __attribute__((ext_vector_type(8)))  float  v8f;

union Frag {
    unsigned u[8];
    uint4    q[2];
    v16bf    v;
};

// float -> bf16 (round to nearest even)
static __device__ __forceinline__ unsigned short f2bf(float x) {
    union { float f; unsigned u; } c;
    c.f = x;
    unsigned r = c.u + 0x7FFFu + ((c.u >> 16) & 1u);
    return (unsigned short)(r >> 16);
}

#define BATCH 16384
#define INF   512
#define OUTF  512
#define NG    1024          // combined gate width (z|h, permuted)
#define NT    (NG / 16)     // 64 N-tiles
#define KK    (INF / 32)    // 16 K-steps
#define MSUB  4             // M-subtiles per wave in the GEMM (64-row strip)
#define SCAN_WGS 4

// K-pair offsets inside a 32-wide K step for the 16-bit A/B fragment layout:
// lanes 0-15 hold K pairs {0,2,4,6} and {16,18,20,22}; lanes 16-31 are +8.
#define SELBASE(i) ((i) < 4 ? 2 * (i) : 16 + 2 * ((i) - 4))

// Fragment-ready weight layout: dword index ((nt*KK + kk)*32 + lane)*8 + i
// so one B fragment = 8 contiguous dwords per lane (two b128 loads).
#define FRAG_BASE(nt, kk, lane) ((((size_t)(nt) * KK + (kk)) * 32 + (lane)) * 8)

// ---------------------------------------------------------------------------
// Kernel X: convert x (f32) -> xb (bf16, row-major). One pass, memory bound.
// ---------------------------------------------------------------------------
__global__ __launch_bounds__(256) void convert_x_kernel(
    const float* __restrict__ x, unsigned short* __restrict__ xb)
{
    int idx = blockIdx.x * blockDim.x + threadIdx.x;   // 0 .. BATCH*INF-1
    xb[idx] = f2bf(x[idx]);
}

// ---------------------------------------------------------------------------
// Kernel P: build fragment-swizzled bf16 weights WcP/UcP with the
// per-workgroup column permutation (WG g owns z-cols [128g,128g+128) then
// h-cols [128g,128g+128)), the combined bias, and reset the barrier counter.
// Each thread emits one packed dword (a K-pair for one lane of one fragment).
// ---------------------------------------------------------------------------
__global__ __launch_bounds__(256) void pack_kernel(
    const float* __restrict__ Wz, const float* __restrict__ Uz,
    const float* __restrict__ bz, const float* __restrict__ Wh,
    const float* __restrict__ Uh, const float* __restrict__ bh,
    unsigned* __restrict__ WcP, unsigned* __restrict__ UcP,
    float* __restrict__ biasc, unsigned* __restrict__ cnt)
{
    int idx = blockIdx.x * blockDim.x + threadIdx.x;   // 0 .. NT*KK*32*8-1
    if (idx == 0) *cnt = 0u;

    const int i    = idx & 7;
    const int lane = (idx >> 3) & 31;
    const int kk   = (idx >> 8) & (KK - 1);
    const int nt   = idx >> 12;                        // 0..63

    const int halfSel = (lane & 16) ? 8 : 0;
    const int k       = (kk << 5) + SELBASE(i) + halfSel;   // K index (pair start)
    const int ncol    = (nt << 4) + (lane & 15);            // packed column

    // packed column -> source column (z/h interleaved by owning workgroup)
    const int g      = ncol >> 8;
    const int c      = ncol & 255;
    const int srccol = (g << 7) + (c & 127);
    const bool isZ   = (c < 128);

    const float* W = isZ ? Wz : Wh;
    const float* U = isZ ? Uz : Uh;

    unsigned wlo = f2bf(W[(size_t)k * OUTF + srccol]);
    unsigned whi = f2bf(W[(size_t)(k + 1) * OUTF + srccol]);
    unsigned ulo = f2bf(U[(size_t)k * OUTF + srccol]);
    unsigned uhi = f2bf(U[(size_t)(k + 1) * OUTF + srccol]);
    WcP[idx] = wlo | (whi << 16);
    UcP[idx] = ulo | (uhi << 16);

    if (idx < NG) {
        const int g2 = idx >> 8, c2 = idx & 255;
        const int sc = (g2 << 7) + (c2 & 127);
        biasc[idx] = (c2 < 128) ? bz[sc] : bh[sc];
    }
}

// ---------------------------------------------------------------------------
// Kernel G: G = x @ [Wz|Wh] + bias  (M=16384, N=1024, K=512) via
// v_wmma_f32_16x16x32_bf16. Each wave computes a 64x16 strip (4 M-subtiles
// sharing one B fragment per K-step) -> 1.25 KB of L2 traffic per WMMA
// instead of 2 KB. Inner loop is pure b128 loads + wmma.
// ---------------------------------------------------------------------------
__global__ __launch_bounds__(256) void gemm_kernel(
    const unsigned short* __restrict__ xb, const unsigned* __restrict__ WcP,
    const float* __restrict__ biasc, float* __restrict__ G)
{
    const int lane  = threadIdx.x & 31;
    const int wave  = threadIdx.x >> 5;
    const int mbase = blockIdx.x * (MSUB * 16);           // 64-row strip base
    const int nt    = blockIdx.y * 8 + wave;              // N tile index
    const int hs    = (lane & 16) ? 8 : 0;                // bf16-element shift
    const int arow  = mbase + (lane & 15);

    const unsigned short* xrow  = xb + (size_t)arow * INF;   // subtile s: +16*s rows
    const unsigned*       wbase = WcP + FRAG_BASE(nt, 0, lane);

    v8f acc[MSUB];
#pragma unroll
    for (int s = 0; s < MSUB; ++s) acc[s] = (v8f){};

#pragma unroll
    for (int kk = 0; kk < KK; ++kk) {
        const int kb = kk << 5;
        Frag b;
        b.q[0] = *(const uint4*)(wbase + (size_t)kk * 256);
        b.q[1] = *(const uint4*)(wbase + (size_t)kk * 256 + 4);
        Frag a[MSUB];
#pragma unroll
        for (int s = 0; s < MSUB; ++s) {
            const unsigned short* xr = xrow + (size_t)(16 * s) * INF;
            a[s].q[0] = *(const uint4*)(xr + kb + hs);        // K 0..7  (+8 hi lanes)
            a[s].q[1] = *(const uint4*)(xr + kb + 16 + hs);   // K 16..23 (+8 hi lanes)
        }
#pragma unroll
        for (int s = 0; s < MSUB; ++s)
            acc[s] = __builtin_amdgcn_wmma_f32_16x16x32_bf16(
                false, a[s].v, false, b.v, (short)0, acc[s], false, false);
    }

    // C/D layout: VGPR r -> M=r (lanes 0-15) / M=r+8 (lanes 16-31), N=lane&15
    const int   rowoff = (lane >> 4) << 3;
    const int   col    = (nt << 4) + (lane & 15);
    const float bv     = biasc[col];
#pragma unroll
    for (int s = 0; s < MSUB; ++s) {
#pragma unroll
        for (int r = 0; r < 8; ++r) {
            int row = mbase + 16 * s + r + rowoff;
            G[(size_t)row * NG + col] = acc[s][r] + bv;
        }
    }
}

// ---------------------------------------------------------------------------
// Kernel S: sequential scan. 4 persistent workgroups x 512 threads (16 waves).
// Each wave pins one 16-column slice of [Uz|Uh] across all K=512 as 16
// persistent bf16 WMMA B-fragments (128 VGPRs/lane). Per step: broadcast
// state m (bf16-packed in LDS, read as two ds_load_b128 per K-step) as the A
// operand, 16 v_wmma per wave, gate, update the locally-owned 128 state
// elements, exchange via an L2 mailbox with a monotonic atomic barrier
// (+ cluster-barrier ISA path, NOP when the dispatch isn't clustered).
// ---------------------------------------------------------------------------
__global__ __launch_bounds__(512) void scan_kernel(
    const float* __restrict__ G, const unsigned* __restrict__ UcP,
    float* __restrict__ out, volatile float* mx, unsigned* cnt)
{
    const int tid  = threadIdx.x;
    const int wg   = blockIdx.x;                    // 0..3
    const int wave = tid >> 5;                      // 0..15
    const int lane = tid & 31;
    const int hs4  = (lane & 16) ? 4 : 0;           // dword shift into K-block
    const int nloc = (wave << 4) + (lane & 15);     // local col 0..255
    const int ncol = (wg << 8) + nloc;              // packed global col
    const int nt   = wg * 16 + wave;                // this wave's N-tile

    __shared__ unsigned mb[256];      // full state m as packed bf16 pairs
    __shared__ float zbuf[128];
    __shared__ float hbuf[128];

    // ---- load persistent B fragments (two b128 per K-step) ----
    const unsigned* ub = UcP + FRAG_BASE(nt, 0, lane);
    Frag bfr[KK];
#pragma unroll
    for (int kk = 0; kk < KK; ++kk) {
        bfr[kk].q[0] = *(const uint4*)(ub + (size_t)kk * 256);
        bfr[kk].q[1] = *(const uint4*)(ub + (size_t)kk * 256 + 4);
    }

    if (tid < 256) mb[tid] = 0u;      // m0 = 0
    float mreg = 0.0f;                // tid<128 owns m[wg*128 + tid]
    __syncthreads();

    for (int t = 0; t < BATCH; ++t) {
        if (lane == 0)
            __builtin_prefetch(G + (size_t)(t + 1) * NG + (wg << 8) + (wave << 4), 0, 1);

        // ---- two matvecs as WMMA with broadcast-m A operand ----
        v8f acc = {};
#pragma unroll
        for (int kk = 0; kk < KK; ++kk) {
            Frag a;
            a.q[0] = *(const uint4*)&mb[(kk << 4) + hs4];       // pairs 0..3 (+4 hi)
            a.q[1] = *(const uint4*)&mb[(kk << 4) + 8 + hs4];   // pairs 8..11 (+4 hi)
            acc = __builtin_amdgcn_wmma_f32_16x16x32_bf16(
                false, a.v, false, bfr[kk].v, (short)0, acc, false, false);
        }

        // row 0 of D lives in acc[0] on lanes 0..15 (M=0, N=lane)
        if (lane < 16) {
            float pre = acc[0] + G[(size_t)t * NG + ncol];
            if (nloc < 128) zbuf[nloc] = 1.0f / (1.0f + __expf(-pre));
            else            hbuf[nloc - 128] = tanhf(pre);
        }
        __syncthreads();

        const int par = t & 1;
        if (tid < 128) {
            float z  = zbuf[tid];
            float h  = hbuf[tid];
            float mn = fmaf(z, h - mreg, mreg);      // (1-z)*m + z*h
            mreg = mn;
            int j = (wg << 7) + tid;
            out[(size_t)t * OUTF + j] = mn;
            mx[par * OUTF + j] = mn;                  // L2 mailbox (release below)
        }
        __syncthreads();

        // ---- cross-workgroup barrier: monotonic counter in L2 ----
        if (tid == 0) {
            __threadfence();
            atomicAdd(cnt, 1u);
            const unsigned target = (unsigned)SCAN_WGS * (unsigned)(t + 1);
            while (atomicAdd(cnt, 0u) < target)
                __builtin_amdgcn_s_sleep(2);
        }
        __syncthreads();
        __builtin_amdgcn_s_cluster_barrier();  // s_barrier_signal/wait -3 (NOP unclustered)
        __threadfence();

        // ---- gather full new state, repack to bf16 for next A operand ----
        if (tid < 256) {
            float a0 = mx[par * OUTF + 2 * tid];
            float a1 = mx[par * OUTF + 2 * tid + 1];
            mb[tid] = (unsigned)f2bf(a0) | ((unsigned)f2bf(a1) << 16);
        }
        __syncthreads();
    }
}

// ---------------------------------------------------------------------------
// Launch
// ---------------------------------------------------------------------------
extern "C" void kernel_launch(void* const* d_in, const int* in_sizes, int n_in,
                              void* d_out, int out_size, void* d_ws, size_t ws_size,
                              hipStream_t stream) {
    const float* x  = (const float*)d_in[0];
    const float* Wz = (const float*)d_in[1];
    const float* Uz = (const float*)d_in[2];
    const float* bz = (const float*)d_in[3];
    const float* Wh = (const float*)d_in[4];
    const float* Uh = (const float*)d_in[5];
    const float* bh = (const float*)d_in[6];
    float* out = (float*)d_out;

    // workspace layout (~82 MB)
    float*          G     = (float*)d_ws;                         // 16384x1024 f32
    unsigned*       WcP   = (unsigned*)(G + (size_t)BATCH * NG);  // 1 MB
    unsigned*       UcP   = WcP + (size_t)NT * KK * 32 * 8;       // 1 MB
    float*          biasc = (float*)(UcP + (size_t)NT * KK * 32 * 8);
    float*          mx    = biasc + NG;                           // 2x512 mailbox
    unsigned*       cnt   = (unsigned*)(mx + NG);
    unsigned short* xb    = (unsigned short*)(cnt + 64);          // 16 MB, aligned

    convert_x_kernel<<<(BATCH * INF) / 256, 256, 0, stream>>>(x, xb);

    pack_kernel<<<(NT * KK * 32 * 8) / 256, 256, 0, stream>>>(
        Wz, Uz, bz, Wh, Uh, bh, WcP, UcP, biasc, cnt);

    gemm_kernel<<<dim3(BATCH / (MSUB * 16), NG / 128), 256, 0, stream>>>(xb, WcP, biasc, G);

    scan_kernel<<<SCAN_WGS, 512, 0, stream>>>(G, UcP, out, mx, cnt);
}
